// SparseAttention_37271726194955
// MI455X (gfx1250) — matrix operand order repaired and checked by reading, not compiled, not run on hardware
//
#include <hip/hip_runtime.h>

// ---------------------------------------------------------------------------
// Windowed multi-head attention for MI455X (gfx1250), wave32 + WMMA bf16.
//   B=4, N=4096, DIM=1024, H=16, HD=64, W=16, n=256, SCALE=1/8
// d_out = [out f32 (16777216)] ++ [attn f32 (67108864)]
// Roofline: 172 GFLOP vs 0.42 GB mandatory traffic (268 MB attn write) ->
// bf16 WMMA (16x16x32, f32 accum) keeps compute at/below the ~18us HBM floor.
// ---------------------------------------------------------------------------

typedef unsigned short u16;
typedef __attribute__((ext_vector_type(16))) __bf16 bf16x16;
typedef __attribute__((ext_vector_type(8)))  float  f32x8;
typedef __attribute__((__vector_size__(16))) int v4i;

struct alignas(16) U128 { unsigned x, y, z, w; };

union OpPack {          // one WMMA 16-bit operand: 16 bf16 = 32B = 2 x b128
    U128    u[2];
    bf16x16 v;
};
union Pack8 {           // 8 bf16 lanes of one b128
    U128 u;
    u16  s[8];
};

__device__ __forceinline__ u16 f2bf(float f) {
    union { float f; unsigned u; } c; c.f = f;
    unsigned r = c.u + 0x7FFFu + ((c.u >> 16) & 1u);   // round-to-nearest-even
    return (u16)(r >> 16);
}

#define WMMA_BF16(A, B, C) \
    __builtin_amdgcn_wmma_f32_16x16x32_bf16(false, (A), false, (B), (short)0, (C), false, false)

// Async global->LDS staging (ASYNCcnt path). Probe round 2 confirmed the
// builtin exists with params (v4i __device__* src, v4i __shared__* dst, Ii, Ii).
#if defined(__gfx1250__) && \
    __has_builtin(__builtin_amdgcn_global_load_async_to_lds_b128) && \
    __has_builtin(__builtin_amdgcn_s_wait_asynccnt)
#define HAVE_ASYNC_LDS 1
typedef __attribute__((address_space(1))) v4i GV4;   // global (AS1)
typedef __attribute__((address_space(3))) v4i LV4;   // LDS (AS3)
#else
#define HAVE_ASYNC_LDS 0
#endif

// ---------------------------------------------------------------------------
// f32 -> bf16 conversion
// ---------------------------------------------------------------------------
__global__ void cvt_bf16_kernel(const float* __restrict__ in,
                                u16* __restrict__ out, int n) {
    int i = blockIdx.x * blockDim.x + threadIdx.x;
    if (i < n) out[i] = f2bf(in[i]);
}

// ---------------------------------------------------------------------------
// C[M,1024] = A[M,1024](bf16) @ W[1024,1024](bf16)^T + bias
//   mode 0: store bf16 scattered into blocked [b,h,w,i,hd] layout (q/k/v)
//   mode 1: store f32 row-major [M,1024] (final projection)
// block = 256 thr (8 waves); wave -> 16 rows x 64 cols; grid = (M/128, 16)
// K-loop software-pipelined (double-buffered operands) so WMMA overlaps the
// next step's global_load_b128s instead of s_wait_loadcnt 0 each step.
// ---------------------------------------------------------------------------
__global__ __launch_bounds__(256) void gemm_bf16_wmma(
    const u16* __restrict__ A, const u16* __restrict__ W,
    const float* __restrict__ bias,
    u16* __restrict__ outB, float* __restrict__ outF, int mode)
{
    const int wave   = threadIdx.x >> 5;
    const int lane   = threadIdx.x & 31;
    const int lanelo = lane & 15;
    const int hi     = (lane >> 4) & 1;
    const int aoff   = hi ? 8 : 0;    // A operand K sub-offset (ISA 16-bit A layout)
    const int boff   = hi ? 16 : 0;   // K offset used only by the W prefetch; B loads use aoff

    const int rbase = blockIdx.x * 128 + wave * 16;
    const int cbase = blockIdx.y * 64;

    const U128* Arow = (const U128*)(A + (size_t)(rbase + lanelo) * 1024);
    const U128* Wrow[4];
#pragma unroll
    for (int ct = 0; ct < 4; ++ct)
        Wrow[ct] = (const U128*)(W + (size_t)(cbase + ct * 16 + lanelo) * 1024);

    auto loadA = [&](OpPack& a, int kb) {
        const int ea = (kb + aoff) >> 3;       // b128 index (8 bf16 each)
        a.u[0] = Arow[ea];
        a.u[1] = Arow[ea + 2];
    };
    auto loadB = [&](OpPack* bb, int kb) {
        const int eb = (kb + aoff) >> 3;
#pragma unroll
        for (int ct = 0; ct < 4; ++ct) {
            bb[ct].u[0] = Wrow[ct][eb];
            bb[ct].u[1] = Wrow[ct][eb + 2];
        }
    };

    f32x8 zero = {0.f,0.f,0.f,0.f,0.f,0.f,0.f,0.f};
    f32x8 acc[4];
#pragma unroll
    for (int ct = 0; ct < 4; ++ct) acc[ct] = zero;

    OpPack a0p, a1p, b0p[4], b1p[4];
    loadA(a0p, 0);
    loadB(b0p, 0);
#pragma unroll 1
    for (int kb = 0; kb < 1024; kb += 64) {
        // stage k+32 while computing k
        loadA(a1p, kb + 32);
        loadB(b1p, kb + 32);
        if (kb + 64 < 1024) {
            __builtin_prefetch(Arow + ((kb + 64 + aoff) >> 3), 0, 1);  // global_prefetch_b8
            __builtin_prefetch(Wrow[0] + ((kb + 64 + boff) >> 3), 0, 1);
        }
#pragma unroll
        for (int ct = 0; ct < 4; ++ct)
            acc[ct] = WMMA_BF16(a0p.v, b0p[ct].v, acc[ct]);
        // stage k+64 while computing k+32
        if (kb + 64 < 1024) {
            loadA(a0p, kb + 64);
            loadB(b0p, kb + 64);
        }
#pragma unroll
        for (int ct = 0; ct < 4; ++ct)
            acc[ct] = WMMA_BF16(a1p.v, b1p[ct].v, acc[ct]);
    }

#pragma unroll
    for (int ct = 0; ct < 4; ++ct) {
        const int col = cbase + ct * 16 + lanelo;
        const float bv = bias[col];
        const int h  = col >> 6;
        const int hd = col & 63;
#pragma unroll
        for (int r = 0; r < 8; ++r) {
            const int   ig  = rbase + r + hi * 8;          // global row
            const float val = acc[ct][r] + bv;
            if (mode == 0) {
                const int b  = ig >> 12;
                const int t  = ig & 4095;
                const int w  = t >> 8;
                const int ii = t & 255;
                const size_t idx =
                    (size_t)(((b * 16 + h) * 16) + w) * 16384 + (size_t)ii * 64 + hd;
                outB[idx] = f2bf(val);
            } else {
                outF[(size_t)ig * 1024 + col] = val;
            }
        }
    }
}

// ---------------------------------------------------------------------------
// Block-local attention: one workgroup per (b,h,w); 256 thr = 8 waves.
// LDS: Q[256x64] + K[256x64] + V^T[64x256] bf16 + per-wave P[16x256] bf16
//      = (3*16384 + 8*4096) u16 = 160 KB
// Q/K staged via GLOBAL_LOAD_ASYNC_TO_LDS_B128 (overlapped with the manual
// V-transpose), completed with s_wait_asynccnt 0 before the barrier.
// ---------------------------------------------------------------------------
__global__ __launch_bounds__(256) void attn_wmma_kernel(
    const u16* __restrict__ qb, const u16* __restrict__ kb,
    const u16* __restrict__ vb, u16* __restrict__ ctx,
    float* __restrict__ attn_out)
{
    extern __shared__ u16 sm[];
    u16* Qs = sm;                 // 16384
    u16* Ks = sm + 16384;         // 16384
    u16* Vt = sm + 32768;         // 16384 (transposed: [d][j])

    const int wave   = threadIdx.x >> 5;
    const int lane   = threadIdx.x & 31;
    const int lanelo = lane & 15;
    const int hi     = (lane >> 4) & 1;
    const int aoff   = hi ? 8 : 0;
    const int boff   = hi ? 16 : 0;

    u16* Pw = sm + 49152 + wave * 4096;   // this wave's 16x256 prob tile

    const int bhw = blockIdx.x;           // (b*16 + h)*16 + w
    const int b_  = bhw >> 8;
    const int h_  = (bhw >> 4) & 15;
    const int w_  = bhw & 15;

    const u16* qsrc = qb + (size_t)bhw * 16384;
    const u16* ksrc = kb + (size_t)bhw * 16384;
    const u16* vsrc = vb + (size_t)bhw * 16384;

    // ---- stage Q, K (async DMA or copy) and V (transpose) into LDS ----
    {
        U128* Qd = (U128*)Qs;  const U128* Qg = (const U128*)qsrc;
        U128* Kd = (U128*)Ks;  const U128* Kg = (const U128*)ksrc;
        const U128* Vg = (const U128*)vsrc;
#if HAVE_ASYNC_LDS
        for (int i = threadIdx.x; i < 2048; i += 256) {
            __builtin_amdgcn_global_load_async_to_lds_b128(
                (GV4*)(Qg + i), (LV4*)(Qd + i), 0, 0);
            __builtin_amdgcn_global_load_async_to_lds_b128(
                (GV4*)(Kg + i), (LV4*)(Kd + i), 0, 0);
        }
#else
        for (int i = threadIdx.x; i < 2048; i += 256) {
            Qd[i] = Qg[i];
            Kd[i] = Kg[i];
        }
#endif
        // V transpose through VGPRs overlaps with the async Q/K DMA
        for (int g = threadIdx.x; g < 2048; g += 256) {
            Pack8 t; t.u = Vg[g];
            const int j  = g >> 3;          // key row
            const int d0 = (g & 7) * 8;     // head-dim base
#pragma unroll
            for (int k = 0; k < 8; ++k)
                Vt[(d0 + k) * 256 + j] = t.s[k];
        }
#if HAVE_ASYNC_LDS
        __builtin_amdgcn_s_wait_asynccnt(0);
#endif
    }
    __syncthreads();

    const U128* Q4 = (const U128*)Qs;
    const U128* K4 = (const U128*)Ks;
    const U128* V4 = (const U128*)Vt;
    const U128* P4 = (const U128*)Pw;

    const f32x8 zero = {0.f,0.f,0.f,0.f,0.f,0.f,0.f,0.f};
    const size_t abase = (size_t)bhw * 65536;   // attn block offset

    for (int rt = wave * 2; rt < wave * 2 + 2; ++rt) {
        // ---- S = (Q K^T) * 1/8 : 16 column tiles of 16, K=64 (2 WMMA each)
        const int arow = (rt * 16 + lanelo) * 64;
        OpPack a0, a1;
        a0.u[0] = Q4[(arow + aoff) >> 3];       a0.u[1] = Q4[((arow + aoff) >> 3) + 2];
        a1.u[0] = Q4[(arow + 32 + aoff) >> 3];  a1.u[1] = Q4[((arow + 32 + aoff) >> 3) + 2];

        f32x8 acc[16];
#pragma unroll
        for (int jt = 0; jt < 16; ++jt) {
            const int brow = (jt * 16 + lanelo) * 64;
            OpPack b0, b1;
            b0.u[0] = K4[(brow + aoff) >> 3];       b0.u[1] = K4[((brow + aoff) >> 3) + 2];
            b1.u[0] = K4[(brow + 32 + aoff) >> 3];  b1.u[1] = K4[((brow + 32 + aoff) >> 3) + 2];
            f32x8 c = zero;
            c = WMMA_BF16(a0.v, b0.v, c);
            c = WMMA_BF16(a1.v, b1.v, c);
#pragma unroll
            for (int r = 0; r < 8; ++r) c[r] *= 0.125f;   // SCALE = HD^-0.5
            acc[jt] = c;
        }

        // ---- softmax over 256 cols (rows live in half-wave lane groups) ----
        float m[8], s[8], inv[8];
#pragma unroll
        for (int r = 0; r < 8; ++r) {
            float mm = acc[0][r];
#pragma unroll
            for (int jt = 1; jt < 16; ++jt) mm = fmaxf(mm, acc[jt][r]);
#pragma unroll
            for (int mk = 1; mk <= 8; mk <<= 1)
                mm = fmaxf(mm, __shfl_xor(mm, mk, 32));   // stays within 16-lane half
            m[r] = mm;
        }
#pragma unroll
        for (int r = 0; r < 8; ++r) s[r] = 0.f;
#pragma unroll
        for (int jt = 0; jt < 16; ++jt)
#pragma unroll
            for (int r = 0; r < 8; ++r) {
                const float p = __expf(acc[jt][r] - m[r]);  // v_exp_f32
                acc[jt][r] = p;
                s[r] += p;
            }
#pragma unroll
        for (int r = 0; r < 8; ++r) {
            float ss = s[r];
#pragma unroll
            for (int mk = 1; mk <= 8; mk <<= 1)
                ss += __shfl_xor(ss, mk, 32);
            inv[r] = 1.0f / ss;
        }

        // ---- write attn probs (f32 -> d_out) and P tile (bf16 -> LDS) ----
#pragma unroll
        for (int jt = 0; jt < 16; ++jt) {
            const int col = jt * 16 + lanelo;
#pragma unroll
            for (int r = 0; r < 8; ++r) {
                const int   row = rt * 16 + r + hi * 8;
                const float p   = acc[jt][r] * inv[r];
                attn_out[abase + (size_t)row * 256 + col] = p;
                Pw[(r + hi * 8) * 256 + col] = f2bf(p);
            }
        }

        // ---- out = P @ V : 4 column tiles of 16, K=256 (8 WMMA each) ----
        f32x8 o[4];
#pragma unroll
        for (int dt = 0; dt < 4; ++dt) o[dt] = zero;
        const int prow = lanelo * 256;
#pragma unroll
        for (int ks = 0; ks < 8; ++ks) {
            const int kbb = ks * 32;
            OpPack a;
            a.u[0] = P4[(prow + kbb + aoff) >> 3];
            a.u[1] = P4[((prow + kbb + aoff) >> 3) + 2];
#pragma unroll
            for (int dt = 0; dt < 4; ++dt) {
                const int drow = (dt * 16 + lanelo) * 256;
                OpPack bop;
                bop.u[0] = V4[(drow + kbb + aoff) >> 3];
                bop.u[1] = V4[((drow + kbb + aoff) >> 3) + 2];
                o[dt] = WMMA_BF16(a.v, bop.v, o[dt]);
            }
        }

        // ---- store context bf16, row-major [b, t, h*64+d] ----
#pragma unroll
        for (int dt = 0; dt < 4; ++dt) {
            const int dim = h_ * 64 + dt * 16 + lanelo;
#pragma unroll
            for (int r = 0; r < 8; ++r) {
                const int row = rt * 16 + r + hi * 8;
                const int t   = w_ * 256 + row;
                ctx[((size_t)(b_ * 4096 + t)) * 1024 + dim] = f2bf(o[dt][r]);
            }
        }
    }
}

// ---------------------------------------------------------------------------
extern "C" void kernel_launch(void* const* d_in, const int* in_sizes, int n_in,
                              void* d_out, int out_size, void* d_ws, size_t ws_size,
                              hipStream_t stream) {
    const float* x  = (const float*)d_in[0];
    const float* Wq = (const float*)d_in[1];
    const float* bq = (const float*)d_in[2];
    const float* Wk = (const float*)d_in[3];
    const float* bk = (const float*)d_in[4];
    const float* Wv = (const float*)d_in[5];
    const float* bv = (const float*)d_in[6];
    const float* Wo = (const float*)d_in[7];
    const float* bo = (const float*)d_in[8];

    const int XN = 16777216;   // B*N*DIM
    const int WN = 1048576;    // DIM*DIM

    u16* xb  = (u16*)d_ws;
    u16* wqb = xb  + XN;
    u16* wkb = wqb + WN;
    u16* wvb = wkb + WN;
    u16* wob = wvb + WN;
    u16* qb  = wob + WN;
    u16* kb  = qb  + XN;
    u16* vb  = kb  + XN;
    u16* ctx = vb  + XN;

    float* out_f    = (float*)d_out;
    float* attn_out = out_f + XN;

    // 1) bf16 conversions
    cvt_bf16_kernel<<<(XN + 255) / 256, 256, 0, stream>>>(x,  xb,  XN);
    cvt_bf16_kernel<<<(WN + 255) / 256, 256, 0, stream>>>(Wq, wqb, WN);
    cvt_bf16_kernel<<<(WN + 255) / 256, 256, 0, stream>>>(Wk, wkb, WN);
    cvt_bf16_kernel<<<(WN + 255) / 256, 256, 0, stream>>>(Wv, wvb, WN);
    cvt_bf16_kernel<<<(WN + 255) / 256, 256, 0, stream>>>(Wo, wob, WN);

    // 2) q/k/v projections, stored blocked bf16
    dim3 gg(128, 16);   // M/128, N/64
    gemm_bf16_wmma<<<gg, 256, 0, stream>>>(xb, wqb, bq, qb, nullptr, 0);
    gemm_bf16_wmma<<<gg, 256, 0, stream>>>(xb, wkb, bk, kb, nullptr, 0);
    gemm_bf16_wmma<<<gg, 256, 0, stream>>>(xb, wvb, bv, vb, nullptr, 0);

    // 3) windowed attention (one WG per (b,h,w)); 160 KB dynamic LDS
    attn_wmma_kernel<<<1024, 256, 163840, stream>>>(qb, kb, vb, ctx, attn_out);

    // 4) output projection -> f32 d_out
    gemm_bf16_wmma<<<gg, 256, 0, stream>>>(ctx, wob, bo, nullptr, out_f, 1);
}
